// MultiHeadAttention_25735444038390
// MI455X (gfx1250) — matrix operand order repaired and checked by reading, not compiled, not run on hardware
//
#include <hip/hip_runtime.h>
#include <string.h>

// MI455X / gfx1250, wave32. Matrix op: v_wmma_f32_16x16x32_bf16.
// Pipeline: [GEMM qkv (bf16 wmma)] -> [flash attention per (b,t,h)] -> [GEMM out + bias]
// CDNA5 paths used: v_wmma (bf16), DPP16 cross-lane reductions (no LDS permutes),
// global_load_async_to_lds_b128 + s_wait_asynccnt for LDS staging where no
// conversion/transpose is needed.

typedef __attribute__((ext_vector_type(16))) __bf16 v16bf;
typedef __attribute__((ext_vector_type(8)))  float  v8f;
typedef __attribute__((ext_vector_type(4)))  int    v4i_t;

union FragBF { v16bf v; uint4 u[2]; };
union FragF  { v8f  v; float f[8]; };

#if defined(__has_builtin)
#if __has_builtin(__builtin_amdgcn_global_load_async_to_lds_b128) && \
    __has_builtin(__builtin_amdgcn_s_wait_asynccnt)
#define HAVE_ASYNC_LDS 1
#endif
#endif
#ifndef HAVE_ASYNC_LDS
#define HAVE_ASYNC_LDS 0
#endif

// 16-byte global->LDS copy; async (ASYNCcnt) when available.
__device__ __forceinline__ void copy16_to_lds(const unsigned short* g, unsigned short* l) {
#if HAVE_ASYNC_LDS
  __builtin_amdgcn_global_load_async_to_lds_b128(
      (__attribute__((address_space(1))) v4i_t*)(uintptr_t)g,
      (__attribute__((address_space(3))) v4i_t*)(uintptr_t)l, 0, 0);
#else
  *(uint4*)l = *(const uint4*)g;
#endif
}

__device__ __forceinline__ void wait_async_copies() {
#if HAVE_ASYNC_LDS
  __builtin_amdgcn_s_wait_asynccnt(0);
#endif
}

__device__ __forceinline__ unsigned short f2bf(float x) {
  unsigned int u = __float_as_uint(x);
  unsigned int r = 0x7FFFu + ((u >> 16) & 1u);   // round-to-nearest-even
  return (unsigned short)((u + r) >> 16);
}

// ---- DPP16 16-lane reductions (rows never cross the 16-lane DPP row) ----
template <int CTRL>
__device__ __forceinline__ float dpp_mov(float x) {
  return __int_as_float(
      __builtin_amdgcn_update_dpp(0, __float_as_int(x), CTRL, 0xf, 0xf, true));
}
__device__ __forceinline__ float row_max16(float x) {
  x = fmaxf(x, dpp_mov<0x0B1>(x));  // quad_perm(1,0,3,2)  : xor 1
  x = fmaxf(x, dpp_mov<0x04E>(x));  // quad_perm(2,3,0,1)  : xor 2
  x = fmaxf(x, dpp_mov<0x141>(x));  // row_half_mirror     : combine quads in 8
  x = fmaxf(x, dpp_mov<0x140>(x));  // row_mirror          : combine 8s in 16
  return x;
}
__device__ __forceinline__ float row_sum16(float x) {
  x += dpp_mov<0x0B1>(x);
  x += dpp_mov<0x04E>(x);
  x += dpp_mov<0x141>(x);
  x += dpp_mov<0x140>(x);
  return x;
}

// A-matrix 16x32 bf16 fragment (ISA 7.12.2): lane m = l&15,
// lanes>=16 offset K by +8; VGPR0-3 hold K {off..off+7}, VGPR4-7 hold {off+16..off+23}.
__device__ __forceinline__ v16bf load_frag_A(const unsigned short* base, int pitch,
                                             int row, int koff, int lane) {
  FragBF f;
  const unsigned short* p = base + (size_t)(row + (lane & 15)) * pitch
                                 + koff + ((lane >> 4) << 3);
  f.u[0] = *(const uint4*)(p);
  f.u[1] = *(const uint4*)(p + 16);
  return f.v;
}

// B-matrix 32x16 bf16 fragment: lane n = l&15, lanes>=16 offset K by +8;
// VGPR0-3 hold K {off..off+7}, VGPR4-7 hold {off+16..off+23}. Source buffer is n-major [n][k].
__device__ __forceinline__ v16bf load_frag_B(const unsigned short* base, int pitch,
                                             int ncol, int koff, int lane) {
  FragBF f;
  const unsigned short* p = base + (size_t)(ncol + (lane & 15)) * pitch
                                 + koff + ((lane >> 4) << 3);
  f.u[0] = *(const uint4*)(p);
  f.u[1] = *(const uint4*)(p + 16);
  return f.v;
}

__device__ __forceinline__ v8f wmma_bf16(v16bf a, v16bf b, v8f c) {
  return __builtin_amdgcn_wmma_f32_16x16x32_bf16(false, a, false, b, (short)0, c,
                                                 false, false);
}

// ---------------------------------------------------------------------------
// GEMM: [M x 512] @ [512 x 512] ; M = 32768. Tile 128x128 per workgroup,
// 8 waves x (32x64), K staged in 64-wide LDS chunks -> 16 WMMA per wave/stage.
// A_F32: A is fp32 (converted to bf16 on the tile-load path) else bf16 (async).
// OUT_F32: write fp32 + bias, else write bf16.
// ---------------------------------------------------------------------------
template <bool A_F32, bool OUT_F32>
__global__ __launch_bounds__(256)
void gemm_tile_kernel(const void* __restrict__ Ag,
                      const float* __restrict__ W0,
                      const float* __restrict__ W1,
                      const float* __restrict__ W2,
                      const float* __restrict__ bias,
                      void* __restrict__ O0, void* __restrict__ O1,
                      void* __restrict__ O2) {
  constexpr int K = 512, N = 512;
  __shared__ __align__(16) unsigned short At[128][72];  // [m][k], pitch 72
  __shared__ __align__(16) unsigned short Bt[128][72];  // [n][k] (transposed W tile)

  const int tid  = threadIdx.x;
  const int lane = tid & 31;
  const int wave = tid >> 5;
  const int m0 = blockIdx.x * 128;
  const int n0 = blockIdx.y * 128;
  const float* W = (blockIdx.z == 0) ? W0 : (blockIdx.z == 1 ? W1 : W2);
  void* Out      = (blockIdx.z == 0) ? O0 : (blockIdx.z == 1 ? O1 : O2);

  const int wm = (wave & 3) * 32;   // 4 waves along M: 0,32,64,96
  const int wn = (wave >> 2) * 64;  // 2 waves along N: 0,64

  FragF acc[2][4];
#pragma unroll
  for (int i = 0; i < 2; ++i)
#pragma unroll
    for (int j = 0; j < 4; ++j)
#pragma unroll
      for (int v = 0; v < 8; ++v) acc[i][j].f[v] = 0.0f;

  for (int kc = 0; kc < K; kc += 64) {
    // ---- stage A tile 128x64 (2 threads per row, 32 elems each) ----
    {
      const int row = tid >> 1;
      const int cs  = (tid & 1) * 32;
      if (A_F32) {
        const float* src = (const float*)Ag + (size_t)(m0 + row) * K + kc + cs;
        const float4* s4 = (const float4*)src;
        unsigned short t16[32];
#pragma unroll
        for (int q = 0; q < 8; ++q) {
          float4 f = s4[q];
          t16[q * 4 + 0] = f2bf(f.x); t16[q * 4 + 1] = f2bf(f.y);
          t16[q * 4 + 2] = f2bf(f.z); t16[q * 4 + 3] = f2bf(f.w);
        }
#pragma unroll
        for (int q = 0; q < 4; ++q)
          *(uint4*)&At[row][cs + q * 8] = *(const uint4*)&t16[q * 8];
      } else {
        const unsigned short* src =
            (const unsigned short*)Ag + (size_t)(m0 + row) * K + kc + cs;
#pragma unroll
        for (int q = 0; q < 4; ++q)
          copy16_to_lds(src + q * 8, &At[row][cs + q * 8]);
      }
    }
    // ---- stage B tile 64x128, transposed to [n][k] (4 threads per k-row) ----
    {
      const int kr  = tid >> 2;        // 0..63
      const int c32 = (tid & 3) * 32;  // 0,32,64,96
      const float* src = W + (size_t)(kc + kr) * N + n0 + c32;
      const float4* s4 = (const float4*)src;
#pragma unroll
      for (int q = 0; q < 8; ++q) {
        float4 f = s4[q];
        Bt[c32 + q * 4 + 0][kr] = f2bf(f.x);
        Bt[c32 + q * 4 + 1][kr] = f2bf(f.y);
        Bt[c32 + q * 4 + 2][kr] = f2bf(f.z);
        Bt[c32 + q * 4 + 3][kr] = f2bf(f.w);
      }
    }
    if constexpr (!A_F32) wait_async_copies();
    __syncthreads();

#pragma unroll
    for (int ks = 0; ks < 2; ++ks) {
      v16bf af[2], bf_[4];
#pragma unroll
      for (int i = 0; i < 2; ++i)
        af[i] = load_frag_A(&At[0][0], 72, wm + i * 16, ks * 32, lane);
#pragma unroll
      for (int j = 0; j < 4; ++j)
        bf_[j] = load_frag_B(&Bt[0][0], 72, wn + j * 16, ks * 32, lane);
#pragma unroll
      for (int i = 0; i < 2; ++i)
#pragma unroll
        for (int j = 0; j < 4; ++j)
          acc[i][j].v = wmma_bf16(af[i], bf_[j], acc[i][j].v);
    }
    __syncthreads();
  }

  // ---- epilogue: C layout row = v + 8*(lane>=16) ----
  const int mbase = m0 + wm;
  const int rhi   = (lane >> 4) << 3;
#pragma unroll
  for (int i = 0; i < 2; ++i)
#pragma unroll
    for (int j = 0; j < 4; ++j) {
      const int col = n0 + wn + j * 16 + (lane & 15);
      float bval = 0.0f;
      if (OUT_F32) bval = bias[col];
#pragma unroll
      for (int v = 0; v < 8; ++v) {
        const int rowm = mbase + i * 16 + v + rhi;
        if (OUT_F32)
          ((float*)Out)[(size_t)rowm * N + col] = acc[i][j].f[v] + bval;
        else
          ((unsigned short*)Out)[(size_t)rowm * N + col] = f2bf(acc[i][j].f[v]);
      }
    }
}

// ---------------------------------------------------------------------------
// Flash attention: one workgroup per (b*t, head). n=256 tokens, HEAD_DIM=64.
// Each of 8 waves owns 32 query rows; streaming softmax over key blocks of 32;
// K/V^T staged in LDS 128 keys at a time (static LDS < 64 KB). K staging is
// async-to-LDS; softmax row stats via DPP16 (no LDS permute traffic).
// ---------------------------------------------------------------------------
__global__ __launch_bounds__(256)
void attn_kernel(const unsigned short* __restrict__ Qg,
                 const unsigned short* __restrict__ Kg,
                 const unsigned short* __restrict__ Vg,
                 unsigned short* __restrict__ Ctx) {
  constexpr int DM = 512;
  __shared__ __align__(16) unsigned short Kl[128][72];    // [key][d]   (B for S=QK^T)
  __shared__ __align__(16) unsigned short Vt[64][136];    // [col][key] (B for P@V)
  __shared__ __align__(16) unsigned short Ps[8][32][40];  // per-wave P scratch

  const int tid  = threadIdx.x;
  const int lane = tid & 31;
  const int wave = tid >> 5;
  const int bt = blockIdx.x >> 3;
  const int h  = blockIdx.x & 7;
  const size_t base = (size_t)bt * 256 * DM + (size_t)h * 64;

  // Q fragments for this wave's 32 rows, 2 row-groups x 2 K-steps (d=0..31,32..63)
  v16bf Qf[2][2];
  {
    const int rowb = wave * 32 + (lane & 15);
    const int offA = (lane >> 4) << 3;
#pragma unroll
    for (int i = 0; i < 2; ++i)
#pragma unroll
      for (int ks = 0; ks < 2; ++ks) {
        const unsigned short* p =
            Qg + base + (size_t)(rowb + i * 16) * DM + ks * 32 + offA;
        FragBF f;
        f.u[0] = *(const uint4*)(p);
        f.u[1] = *(const uint4*)(p + 16);
        Qf[i][ks] = f.v;
      }
  }

  FragF O[2][4];
  float mrun[2][8], lrun[2][8];
#pragma unroll
  for (int i = 0; i < 2; ++i)
#pragma unroll
    for (int v = 0; v < 8; ++v) {
      mrun[i][v] = -1e30f;
      lrun[i][v] = 0.0f;
#pragma unroll
      for (int g = 0; g < 4; ++g) O[i][g].f[v] = 0.0f;
    }

  for (int s = 0; s < 2; ++s) {      // key stages of 128 tokens
    __syncthreads();                 // prior stage fully consumed
    {
      const int tk   = tid >> 1;            // 0..127 local key
      const int half = (tid & 1) * 32;      // d-columns 0..31 or 32..63
      const unsigned short* kp = Kg + base + (size_t)(s * 128 + tk) * DM + half;
      const unsigned short* vp = Vg + base + (size_t)(s * 128 + tk) * DM + half;
#pragma unroll
      for (int q = 0; q < 4; ++q)
        copy16_to_lds(kp + q * 8, &Kl[tk][half + q * 8]);   // async path
      uint4 vv[4];
#pragma unroll
      for (int q = 0; q < 4; ++q) vv[q] = ((const uint4*)vp)[q];
      const unsigned short* vs = (const unsigned short*)vv;
#pragma unroll
      for (int c = 0; c < 32; ++c) Vt[half + c][tk] = vs[c];  // transpose to [col][key]
    }
    wait_async_copies();
    __syncthreads();

    for (int kb = 0; kb < 4; ++kb) {  // 32 keys per block
      // S = Q K^T  (2 row-groups x 2 key-col-groups)
      FragF S[2][2];
#pragma unroll
      for (int i = 0; i < 2; ++i)
#pragma unroll
        for (int j = 0; j < 2; ++j)
#pragma unroll
          for (int v = 0; v < 8; ++v) S[i][j].f[v] = 0.0f;
#pragma unroll
      for (int j = 0; j < 2; ++j)
#pragma unroll
        for (int ks = 0; ks < 2; ++ks) {
          v16bf kf = load_frag_B(&Kl[0][0], 72, kb * 32 + j * 16, ks * 32, lane);
#pragma unroll
          for (int i = 0; i < 2; ++i)
            S[i][j].v = wmma_bf16(Qf[i][ks], kf, S[i][j].v);
        }

      // streaming softmax update (row = v + 8*(lane>=16); DPP16 reductions)
      const int rhi = (lane >> 4) << 3;
#pragma unroll
      for (int i = 0; i < 2; ++i)
#pragma unroll
        for (int v = 0; v < 8; ++v) {
          float s0 = S[i][0].f[v] * 0.125f;   // 1/sqrt(64)
          float s1 = S[i][1].f[v] * 0.125f;
          float t  = row_max16(fmaxf(s0, s1));
          const float mo = mrun[i][v];
          const float mn = fmaxf(mo, t);
          const float al = __expf(mo - mn);
          const float p0 = __expf(s0 - mn);
          const float p1 = __expf(s1 - mn);
          const float rs = row_sum16(p0 + p1);
          lrun[i][v] = lrun[i][v] * al + rs;
          mrun[i][v] = mn;
          const int prow = i * 16 + v + rhi;
          Ps[wave][prow][(lane & 15)]      = f2bf(p0);
          Ps[wave][prow][16 + (lane & 15)] = f2bf(p1);
#pragma unroll
          for (int g = 0; g < 4; ++g) O[i][g].f[v] *= al;
        }

      // O += P @ V   (A = P from per-wave scratch, B = V^T slice)
      v16bf vf[4];
#pragma unroll
      for (int g = 0; g < 4; ++g)
        vf[g] = load_frag_B(&Vt[0][0], 136, g * 16, kb * 32, lane);
#pragma unroll
      for (int i = 0; i < 2; ++i) {
        v16bf pf = load_frag_A(&Ps[wave][0][0], 40, i * 16, 0, lane);
#pragma unroll
        for (int g = 0; g < 4; ++g)
          O[i][g].v = wmma_bf16(pf, vf[g], O[i][g].v);
      }
    }
  }

  // finalize: divide by row sums, store ctx (bf16)
  const int rhi = (lane >> 4) << 3;
#pragma unroll
  for (int i = 0; i < 2; ++i)
#pragma unroll
    for (int v = 0; v < 8; ++v) {
      const float inv = 1.0f / lrun[i][v];
      const int row = wave * 32 + i * 16 + v + rhi;
#pragma unroll
      for (int g = 0; g < 4; ++g) {
        const int col = g * 16 + (lane & 15);
        Ctx[base + (size_t)row * DM + col] = f2bf(O[i][g].f[v] * inv);
      }
    }
}

// ---------------------------------------------------------------------------
extern "C" void kernel_launch(void* const* d_in, const int* in_sizes, int n_in,
                              void* d_out, int out_size, void* d_ws, size_t ws_size,
                              hipStream_t stream) {
  (void)in_sizes; (void)n_in; (void)out_size; (void)ws_size;
  const float* x  = (const float*)d_in[0];
  const float* Wq = (const float*)d_in[1];
  const float* Wk = (const float*)d_in[2];
  const float* Wv = (const float*)d_in[3];
  const float* Wo = (const float*)d_in[4];
  const float* bo = (const float*)d_in[5];
  float* out = (float*)d_out;

  const size_t ROWS = 4ull * 32 * 256;  // 32768 tokens total
  unsigned short* qb = (unsigned short*)d_ws;
  unsigned short* kb = qb + ROWS * 512;
  unsigned short* vb = kb + ROWS * 512;
  unsigned short* cb = vb + ROWS * 512;   // total ws use: 128 MB

  dim3 blk(256);
  // QKV projections: x(fp32) -> q/k/v (bf16). z selects Wq/Wk/Wv; x tiles L2-resident.
  gemm_tile_kernel<true, false><<<dim3(256, 4, 3), blk, 0, stream>>>(
      x, Wq, Wk, Wv, nullptr, qb, kb, vb);
  // Attention: 1024 = (4*32) * 8 heads
  attn_kernel<<<dim3(1024), blk, 0, stream>>>(qb, kb, vb, cb);
  // Output projection: ctx(bf16) @ Wo + bo -> fp32
  gemm_tile_kernel<false, true><<<dim3(256, 4, 1), blk, 0, stream>>>(
      cb, Wo, Wo, Wo, bo, out, out, out);
}
